// LocalAttention_11450382812006
// MI455X (gfx1250) — matrix operand order repaired and checked by reading, not compiled, not run on hardware
//
#include <hip/hip_runtime.h>
#include <hip/hip_bf16.h>
#include <math.h>

// ---------------------------------------------------------------------------
// Local attention for MI455X (gfx1250, wave32, WMMA f32_16x16x32_f16).
//
// Roofline: ~40 GFLOP, all operands (<90MB) L2-resident (192MB) -> latency /
// issue bound, not HBM. Strategy:
//  * f16 WMMA with f32 accumulation.
//  * Single-wave workgroups for attention (1024 WGs -> full WGP spread).
//  * Low accumulator pressure; fully unrolled, double-buffered GEMM loops.
//  * __launch_bounds__(N,1): occupancy is ~1 wave/SIMD anyway, so free the
//    register budget for load pipelining.
//  * sched_group_barrier "8 VMEM-read, then WMMA" per iteration with one
//    extra VMEM group up front -> steady state issues loads(i+1) before
//    wmma(i): each WMMA waits with 8 b128 loads still in flight, not 0.
//
// Operand layouts per CDNA5 ISA 7.12.2:
//   A (16x32 f16): lane L -> row M=L%16; two contiguous 16B chunks at
//                  K = 8*(L/16) and K = 16+8*(L/16).
//   B (32x16 f16): lane L -> col N=L%16; two contiguous 16B chunks at
//                  K = 8*(L/16) and K = 16+8*(L/16) (requires N-major storage).
//   C/D (16x16 f32): VGPR r -> M = r + 8*(L/16), N = L%16.
// ---------------------------------------------------------------------------

#define DI __device__ __forceinline__

typedef _Float16 half_t;
typedef __attribute__((ext_vector_type(16))) _Float16 v16h;
typedef __attribute__((ext_vector_type(8)))  _Float16 v8h;
typedef __attribute__((ext_vector_type(4)))  _Float16 v4h;
typedef __attribute__((ext_vector_type(8)))  float    v8f;

constexpr int B_   = 4;
constexpr int S_   = 4096;
constexpr int E_   = 512;
constexpr int PAD_ = 128;
constexpr int SP_  = S_ + 2 * PAD_;   // 4352 padded sequence length
constexpr int BLK  = 128;             // query block size (== PAD)
constexpr int SPB  = SP_ / BLK;       // 34 padded blocks
constexpr int WKEY = 3 * BLK;         // 384 keys in window
constexpr float SCALE = 0.04419417382415922f;  // 1/sqrt(512)

// sched_group_barrier masks
#define SGB_VMEM_READ 0x020
#define SGB_WMMA      0x008

union v16u { v16h v; v8h h[2]; };

// A-operand load from row-major [M][K] f16, top-left at p0 (=(row0,k0)).
DI v16h load_a16(const half_t* p0, int ld, int lane) {
  const half_t* p = p0 + (size_t)(lane & 15) * ld + ((lane >> 4) << 3);
  v16u u;
  u.h[0] = *(const v8h*)(p);        // K = k0 + 8h .. +7
  u.h[1] = *(const v8h*)(p + 16);   // K = k0 + 16 + 8h .. +7
  return u.v;
}

// B-operand load from N-major [N][K] f16, top-left at p0 (=(n0,k0)).
DI v16h load_b16(const half_t* p0, int ld, int lane) {
  const half_t* p = p0 + (size_t)(lane & 15) * ld + ((lane >> 4) << 3);
  v16u u;
  u.h[0] = *(const v8h*)(p);        // K = k0 + 8h .. +7
  u.h[1] = *(const v8h*)(p + 16);   // K = k0 + 16 + 8h .. +7
  return u.v;
}

DI v8f wmma_f16(v16h a, v16h b, v8f c) {
  return __builtin_amdgcn_wmma_f32_16x16x32_f16(false, a, false, b,
                                                (short)0, c, false, false);
}

// 3 N-tiles, fully unrolled, double-buffered; NV = VMEM reads pinned per iter.
template <int KEND, int NV>
DI void gemm3_db(const half_t* A, int lda,
                 const half_t* B0, const half_t* B1, const half_t* B2,
                 int ldb, int lane, v8f& c0, v8f& c1, v8f& c2) {
  v16h a  = load_a16(A, lda, lane);
  v16h b0 = load_b16(B0, ldb, lane);
  v16h b1 = load_b16(B1, ldb, lane);
  v16h b2 = load_b16(B2, ldb, lane);
  __builtin_amdgcn_sched_group_barrier(SGB_VMEM_READ, NV, 0);  // prologue
#pragma unroll
  for (int k = 0; k < KEND; k += 32) {
    v16h an = a, n0 = b0, n1 = b1, n2 = b2;
    if (k + 32 < KEND) {
      an = load_a16(A + k + 32, lda, lane);
      n0 = load_b16(B0 + k + 32, ldb, lane);
      n1 = load_b16(B1 + k + 32, ldb, lane);
      n2 = load_b16(B2 + k + 32, ldb, lane);
    }
    c0 = wmma_f16(a, b0, c0);
    c1 = wmma_f16(a, b1, c1);
    c2 = wmma_f16(a, b2, c2);
    a = an; b0 = n0; b1 = n1; b2 = n2;
    __builtin_amdgcn_sched_group_barrier(SGB_VMEM_READ, NV, 0);  // loads(i+1)
    __builtin_amdgcn_sched_group_barrier(SGB_WMMA, 3, 0);        // wmma(i)
  }
}

// 4 N-tiles, fully unrolled, double-buffered; NV = VMEM reads pinned per iter.
template <int KEND, int NV>
DI void gemm4_db(const half_t* A, int lda,
                 const half_t* B0, const half_t* B1,
                 const half_t* B2, const half_t* B3,
                 int ldb, int lane, v8f& c0, v8f& c1, v8f& c2, v8f& c3) {
  v16h a  = load_a16(A, lda, lane);
  v16h b0 = load_b16(B0, ldb, lane);
  v16h b1 = load_b16(B1, ldb, lane);
  v16h b2 = load_b16(B2, ldb, lane);
  v16h b3 = load_b16(B3, ldb, lane);
  __builtin_amdgcn_sched_group_barrier(SGB_VMEM_READ, NV, 0);  // prologue
#pragma unroll
  for (int k = 0; k < KEND; k += 32) {
    v16h an = a, n0 = b0, n1 = b1, n2 = b2, n3 = b3;
    if (k + 32 < KEND) {
      an = load_a16(A + k + 32, lda, lane);
      n0 = load_b16(B0 + k + 32, ldb, lane);
      n1 = load_b16(B1 + k + 32, ldb, lane);
      n2 = load_b16(B2 + k + 32, ldb, lane);
      n3 = load_b16(B3 + k + 32, ldb, lane);
    }
    c0 = wmma_f16(a, b0, c0);
    c1 = wmma_f16(a, b1, c1);
    c2 = wmma_f16(a, b2, c2);
    c3 = wmma_f16(a, b3, c3);
    a = an; b0 = n0; b1 = n1; b2 = n2; b3 = n3;
    __builtin_amdgcn_sched_group_barrier(SGB_VMEM_READ, NV, 0);  // loads(i+1)
    __builtin_amdgcn_sched_group_barrier(SGB_WMMA, 4, 0);        // wmma(i)
  }
}

// --------------------------- prep kernels ----------------------------------

// xh[b][sp][e] = f16(padded x); pad rows are zero (Linear(0) = bias).
__global__ void prep_x_kernel(const float* __restrict__ x, half_t* __restrict__ xh) {
  size_t i = (size_t)blockIdx.x * 256 + threadIdx.x;  // over B*SP*E, exact
  int e = (int)(i % E_);
  size_t t = i / E_;
  int sp = (int)(t % SP_);
  int b  = (int)(t / SP_);
  float v = 0.0f;
  if (sp >= PAD_ && sp < S_ + PAD_)
    v = x[((size_t)b * S_ + (sp - PAD_)) * E_ + e];
  xh[i] = (half_t)v;
}

// WT[w][n][k] = f16(W_w[k][n])  (N-major for the WMMA B-operand)
__global__ void prep_w_kernel(const float* __restrict__ Wq,
                              const float* __restrict__ Wk,
                              const float* __restrict__ Wv,
                              half_t* __restrict__ WT) {
  int i = blockIdx.x * 256 + threadIdx.x;   // over 3*E*E, exact
  int k = i % E_;
  int t = i / E_;
  int n = t % E_;
  int w = t / E_;
  const float* W = (w == 0) ? Wq : (w == 1) ? Wk : Wv;
  WT[i] = (half_t)W[(size_t)k * E_ + n];
}

// --------------------------- projection GEMM -------------------------------
// grid: (E/64, B*SPB, 3[z: q,k,v]); 256 threads = 8 waves, each wave a
// 16(row)x64(col) strip. z==0 (q) skips the pad blocks (blocks 0,33).
__global__ __launch_bounds__(256, 1) void proj_kernel(
    const half_t* __restrict__ xh, const half_t* __restrict__ WT,
    const float* __restrict__ bq, const float* __restrict__ bk,
    const float* __restrict__ bv,
    half_t* __restrict__ qh, half_t* __restrict__ kh, half_t* __restrict__ vth) {
  const int z   = blockIdx.z;
  const int rb  = blockIdx.y;
  const int b   = rb / SPB;
  const int spb = rb % SPB;
  if (z == 0 && (spb == 0 || spb == SPB - 1)) return;  // uniform branch

  const int lane   = threadIdx.x & 31;
  const int w      = threadIdx.x >> 5;
  const int halfid = lane >> 4;
  const int nl     = lane & 15;
  const int row0   = spb * BLK + w * 16;     // padded-seq row base for wave
  const int n0     = blockIdx.x * 64;

  const half_t* Arow = xh + ((size_t)b * SP_ + row0) * E_;
  const half_t* Wz   = WT + (size_t)z * E_ * E_ + (size_t)n0 * E_;
  const float*  bias = (z == 0) ? bq : (z == 1) ? bk : bv;

  v8f c[4] = {};
  gemm4_db<E_, 10>(Arow, E_, Wz, Wz + 16 * E_, Wz + 32 * E_, Wz + 48 * E_,
                   E_, lane, c[0], c[1], c[2], c[3]);

#pragma unroll
  for (int nt = 0; nt < 4; ++nt) {
    const int   n  = n0 + nt * 16 + nl;
    const float bb = bias[n];
#pragma unroll
    for (int r = 0; r < 8; ++r) {
      const int sp  = row0 + r + (halfid << 3);
      const float v = c[nt][r] + bb;
      if (z == 0)
        qh[((size_t)b * S_ + (sp - PAD_)) * E_ + n] = (half_t)v;
      else if (z == 1)
        kh[((size_t)b * SP_ + sp) * E_ + n] = (half_t)v;
      else
        vth[((size_t)b * E_ + n) * SP_ + sp] = (half_t)v;   // transposed
    }
  }
}

// --------------------------- attention -------------------------------------
// grid: (S/16, B); SINGLE-WAVE workgroups (32 threads). Each wave owns 16
// query rows. Score tiles computed 3 at a time (gemm3_db), masked, stashed
// f32 into Slds; softmax from LDS (2 lanes per row); P written f16 into Plds
// (conflict-padded) as the PV A-operand; then P @ V^T via gemm4_db.
__global__ __launch_bounds__(32, 1) void attn_kernel(
    const half_t* __restrict__ qh, const half_t* __restrict__ kh,
    const half_t* __restrict__ vth, const int* __restrict__ mask,
    float* __restrict__ out) {
  constexpr int SSTR = WKEY + 12;  // 396 f32: rows 16B-aligned, banks spread
  constexpr int PSTR = WKEY + 8;   // 392 f16: conflict-free A-operand b128
  __shared__ __align__(16) float  Slds[16][SSTR];  // 25,344 B
  __shared__ __align__(16) half_t Plds[16][PSTR];  // 12,544 B

  const int wq = blockIdx.x;        // 16-row query group: 0..255
  const int b  = blockIdx.y;
  const int nb = wq >> 3;           // 128-row query block (key window base)
  const int wg = wq & 7;            // 16-row group within the block
  const int lane   = threadIdx.x & 31;
  const int halfid = lane >> 4;
  const int nl     = lane & 15;
  const int qrow0  = wq * 16;       // global query row base

  const half_t* Aq = qh + ((size_t)b * S_ + qrow0) * E_;
  const half_t* Bk = kh + ((size_t)b * SP_ + (size_t)nb * BLK) * E_;

  // ---- S = Q @ K^T, 8 groups of 3 key tiles; mask + stash to LDS ----
  for (int tg = 0; tg < 8; ++tg) {
    v8f c[3] = {};
    const half_t* Bt = Bk + (size_t)(tg * 48) * E_;
    gemm3_db<E_, 8>(Aq, E_, Bt, Bt + 16 * E_, Bt + 32 * E_, E_, lane,
                    c[0], c[1], c[2]);
#pragma unroll
    for (int j = 0; j < 3; ++j) {
      const int t   = tg * 3 + j;
      const int klv = t * 16 + nl;               // key idx within 384 window
      const int kp  = nb * BLK + klv;            // padded-seq key position
      const int pm  = (kp < PAD_ || kp >= S_ + PAD_)
                          ? 1
                          : mask[(size_t)b * S_ + (kp - PAD_)];
#pragma unroll
      for (int r = 0; r < 8; ++r) {
        const int ql = wg * 16 + r + (halfid << 3);  // query idx within block
        float v = c[j][r] * SCALE;
        if (pm == 0) v = -1e9f;                      // pad-mask first
        if (klv < ql || klv > ql + 256) v = -__builtin_inff();  // band last
        Slds[r + (halfid << 3)][klv] = v;
      }
    }
  }
  __syncthreads();   // single-wave WG: S_NOP; orders LDS store->load

  // ---- softmax from LDS: 2 lanes per row, 192 f32 each ----
  {
    const int row = lane >> 1, hf = lane & 1;
    float* srow = &Slds[row][hf * 192];
    float m = -__builtin_inff();
    for (int i = 0; i < 192; i += 4) {
      float4 v = *(const float4*)(srow + i);
      m = fmaxf(m, fmaxf(fmaxf(v.x, v.y), fmaxf(v.z, v.w)));
    }
    m = fmaxf(m, __shfl_xor(m, 1));
    float s = 0.0f;
    for (int i = 0; i < 192; i += 4) {
      float4 v = *(const float4*)(srow + i);
      v.x = __expf(v.x - m); v.y = __expf(v.y - m);
      v.z = __expf(v.z - m); v.w = __expf(v.w - m);
      *(float4*)(srow + i) = v;
      s += v.x + v.y + v.z + v.w;
    }
    s += __shfl_xor(s, 1);
    const float inv = 1.0f / s;   // s >= 1 (in-band entries >= -1e9)
    for (int i = 0; i < 192; i += 4) {
      float4 v = *(const float4*)(srow + i);
      v4h h;
      h[0] = (half_t)(v.x * inv); h[1] = (half_t)(v.y * inv);
      h[2] = (half_t)(v.z * inv); h[3] = (half_t)(v.w * inv);
      *(v4h*)(&Plds[row][hf * 192 + i]) = h;
    }
  }
  __syncthreads();   // single-wave WG: S_NOP; orders LDS store->load

  // ---- query-row output mask ----
  float mrow[8];
#pragma unroll
  for (int r = 0; r < 8; ++r)
    mrow[r] = mask[(size_t)b * S_ + qrow0 + r + (halfid << 3)] ? 1.0f : 0.0f;

  // ---- O = P @ V (V transposed: vth[b][e][pos] is B-operand), 64-e chunks --
  for (int ec = 0; ec < 8; ++ec) {
    v8f c[4] = {};
    const half_t* Vb =
        vth + ((size_t)b * E_ + ec * 64) * SP_ + (size_t)nb * BLK;
    gemm4_db<WKEY, 8>(&Plds[0][0], PSTR,
                      Vb, Vb + 16 * SP_, Vb + 32 * SP_, Vb + 48 * SP_,
                      SP_, lane, c[0], c[1], c[2], c[3]);
#pragma unroll
    for (int et = 0; et < 4; ++et) {
      const int e = ec * 64 + et * 16 + nl;
#pragma unroll
      for (int r = 0; r < 8; ++r) {
        const int q = qrow0 + r + (halfid << 3);
        out[((size_t)b * S_ + q) * E_ + e] = c[et][r] * mrow[r];
      }
    }
  }
}

// --------------------------- host launcher ---------------------------------
extern "C" void kernel_launch(void* const* d_in, const int* in_sizes, int n_in,
                              void* d_out, int out_size, void* d_ws, size_t ws_size,
                              hipStream_t stream) {
  const float* x    = (const float*)d_in[0];
  const int*   mask = (const int*)d_in[1];
  const float* Wq   = (const float*)d_in[2];
  const float* bq   = (const float*)d_in[3];
  const float* Wk   = (const float*)d_in[4];
  const float* bk   = (const float*)d_in[5];
  const float* Wv   = (const float*)d_in[6];
  const float* bv   = (const float*)d_in[7];
  float* out = (float*)d_out;

  // Workspace layout (f16), ~72 MB total:
  char*  ws  = (char*)d_ws;
  size_t off = 0;
  half_t* xh  = (half_t*)(ws + off); off += (size_t)B_ * SP_ * E_ * sizeof(half_t);
  half_t* qh  = (half_t*)(ws + off); off += (size_t)B_ * S_  * E_ * sizeof(half_t);
  half_t* kh  = (half_t*)(ws + off); off += (size_t)B_ * SP_ * E_ * sizeof(half_t);
  half_t* vth = (half_t*)(ws + off); off += (size_t)B_ * SP_ * E_ * sizeof(half_t);
  half_t* WT  = (half_t*)(ws + off); off += (size_t)3 * E_ * E_ * sizeof(half_t);

  prep_x_kernel<<<(B_ * SP_ * E_) / 256, 256, 0, stream>>>(x, xh);
  prep_w_kernel<<<(3 * E_ * E_) / 256, 256, 0, stream>>>(Wq, Wk, Wv, WT);
  proj_kernel<<<dim3(E_ / 64, B_ * SPB, 3), 256, 0, stream>>>(
      xh, WT, bq, bk, bv, qh, kh, vth);
  attn_kernel<<<dim3(S_ / 16, B_), 32, 0, stream>>>(qh, kh, vth, mask, out);
}
